// NMTDecoder_13400297963522
// MI455X (gfx1250) — matrix operand order repaired and checked by reading, not compiled, not run on hardware
//
#include <hip/hip_runtime.h>
#include <hip/hip_bf16.h>
#include <math.h>

typedef __attribute__((ext_vector_type(16))) __bf16 v16bf;
typedef __attribute__((ext_vector_type(8)))  __bf16 v8bf;
typedef __attribute__((ext_vector_type(4)))  __bf16 v4bf;
typedef __attribute__((ext_vector_type(8)))  float  v8f;
typedef __attribute__((ext_vector_type(4)))  float  v4f;
typedef __attribute__((ext_vector_type(4)))  unsigned int u32x4;
typedef __attribute__((ext_vector_type(8)))  int i32x8;
typedef __attribute__((ext_vector_type(4)))  int i32x4;

#define V_   32000
#define D_   256
#define H_   512
#define B_   32
#define T_   64
#define KDH  768    // D+H
#define G2H  1024   // 2*H

// ---------------------------------------------------------------------------
// Wout [512][32000] fp32 -> WoutT [32000][512] bf16 (tiled transpose, per call)
// grid = (V/32, H/32), block = 256
// ---------------------------------------------------------------------------
__global__ __launch_bounds__(256)
void wout_transpose_bf16(const float* __restrict__ w, __bf16* __restrict__ o) {
    __shared__ float tile[32][33];
    const int n0 = blockIdx.x * 32;
    const int k0 = blockIdx.y * 32;
    const int tid = threadIdx.x;
    const int r  = tid >> 3;         // 0..31
    const int c4 = (tid & 7) * 4;    // 0,4,...,28

    v4f f = *(const v4f*)(w + (size_t)(k0 + r) * V_ + n0 + c4);
    tile[r][c4 + 0] = f[0]; tile[r][c4 + 1] = f[1];
    tile[r][c4 + 2] = f[2]; tile[r][c4 + 3] = f[3];
    __syncthreads();

    v4bf b;
    b[0] = (__bf16)tile[c4 + 0][r];
    b[1] = (__bf16)tile[c4 + 1][r];
    b[2] = (__bf16)tile[c4 + 2][r];
    b[3] = (__bf16)tile[c4 + 3][r];
    *(v4bf*)(o + (size_t)(n0 + r) * H_ + k0 + c4) = b;
}

// ---------------------------------------------------------------------------
// GRU gate GEMM+sigmoid for one timestep (fp32 VALU; M=32 is too small for WMMA)
// ---------------------------------------------------------------------------
__global__ __launch_bounds__(256)
void gru_gates(const int* __restrict__ wi, const float* __restrict__ hin,
               const float* __restrict__ embW, const float* __restrict__ Wg,
               const float* __restrict__ bgv, float* __restrict__ gates, int t) {
    __shared__ float xh[KDH][8];     // [k][bi]
    const int tid = threadIdx.x;
    const int jt  = blockIdx.x;      // 0..3
    const int bg8 = blockIdx.y;      // 0..3

    for (int idx = tid; idx < KDH * 8; idx += 256) {
        int bi = idx & 7, k = idx >> 3;
        int b = bg8 * 8 + bi;
        float v;
        if (k < D_) v = embW[(size_t)wi[b * T_ + t] * D_ + k];
        else        v = hin[b * H_ + (k - D_)];
        xh[k][bi] = v;
    }
    __syncthreads();

    const int j = jt * 256 + tid;
    float acc[8];
#pragma unroll
    for (int i = 0; i < 8; ++i) acc[i] = 0.f;

#pragma unroll 4
    for (int k = 0; k < KDH; ++k) {
        float w = Wg[(size_t)k * G2H + j];
        v4f a0 = *(const v4f*)&xh[k][0];
        v4f a1 = *(const v4f*)&xh[k][4];
        acc[0] += a0[0] * w; acc[1] += a0[1] * w;
        acc[2] += a0[2] * w; acc[3] += a0[3] * w;
        acc[4] += a1[0] * w; acc[5] += a1[1] * w;
        acc[6] += a1[2] * w; acc[7] += a1[3] * w;
    }

    float bias = bgv[j];
#pragma unroll
    for (int bi = 0; bi < 8; ++bi) {
        int b = bg8 * 8 + bi;
        gates[(size_t)b * G2H + j] = 1.f / (1.f + expf(-(acc[bi] + bias)));
    }
}

// ---------------------------------------------------------------------------
// GRU candidate + state update for one timestep
// ---------------------------------------------------------------------------
__global__ __launch_bounds__(256)
void gru_cand(const int* __restrict__ wi, const int* __restrict__ nw,
              const float* __restrict__ hin, const float* __restrict__ embW,
              const float* __restrict__ gates, const float* __restrict__ Wc,
              const float* __restrict__ bcv, float* __restrict__ hout,
              __bf16* __restrict__ Aout, int t) {
    __shared__ float xrh[KDH][8];    // [k][bi]
    const int tid = threadIdx.x;
    const int it  = blockIdx.x;      // 0..1
    const int bg8 = blockIdx.y;      // 0..3

    for (int idx = tid; idx < KDH * 8; idx += 256) {
        int bi = idx & 7, k = idx >> 3;
        int b = bg8 * 8 + bi;
        float v;
        if (k < D_) v = embW[(size_t)wi[b * T_ + t] * D_ + k];
        else {
            int kk = k - D_;
            v = gates[(size_t)b * G2H + kk] * hin[b * H_ + kk];  // r*h
        }
        xrh[k][bi] = v;
    }
    __syncthreads();

    const int i = it * 256 + tid;
    float acc[8];
#pragma unroll
    for (int q = 0; q < 8; ++q) acc[q] = 0.f;

#pragma unroll 4
    for (int k = 0; k < KDH; ++k) {
        float w = Wc[(size_t)k * H_ + i];
        v4f a0 = *(const v4f*)&xrh[k][0];
        v4f a1 = *(const v4f*)&xrh[k][4];
        acc[0] += a0[0] * w; acc[1] += a0[1] * w;
        acc[2] += a0[2] * w; acc[3] += a0[3] * w;
        acc[4] += a1[0] * w; acc[5] += a1[1] * w;
        acc[6] += a1[2] * w; acc[7] += a1[3] * w;
    }

    float bias = bcv[i];
#pragma unroll
    for (int bi = 0; bi < 8; ++bi) {
        int b = bg8 * 8 + bi;
        float c  = tanhf(acc[bi] + bias);
        float u  = gates[(size_t)b * G2H + H_ + i];
        float h0 = hin[b * H_ + i];
        float hn = u * h0 + (1.f - u) * c;
        float m  = (t < nw[b]) ? 1.f : 0.f;
        Aout[((size_t)t * B_ + b) * H_ + i] = (__bf16)(m * hn);   // masked output row
        hout[b * H_ + i] = m * hn + (1.f - m) * h0;               // carried state
    }
}

// ---------------------------------------------------------------------------
// TDM 2D tile load: [tile_d1 rows][32 bf16] from a row-major tensor with
// row stride `stride0` (elements) into LDS with 80B padded row stride
// (pad_interval = 16 dwords, pad_amount = 4 dwords -> LDS stride 40 bf16).
// ---------------------------------------------------------------------------
__device__ __forceinline__
void tdm_load_2d(unsigned lds_off, const __bf16* gptr,
                 unsigned tensor_d0, unsigned tensor_d1,
                 unsigned tile_d0, unsigned tile_d1, unsigned stride0) {
    unsigned long long ga = (unsigned long long)(uintptr_t)gptr;
    u32x4 g0;
    g0[0] = 1u;                                            // count=1, no gather
    g0[1] = lds_off;                                       // lds_addr
    g0[2] = (unsigned)ga;                                  // global_addr[31:0]
    g0[3] = (unsigned)((ga >> 32) & 0x1FFFFFFu) | (2u << 30);  // addr[56:32], type=2
    i32x8 g1;
    g1[0] = (int)((1u << 16) | (1u << 20) | (3u << 22) | (3u << 25));
           // data_size=1(2B), pad_enable, pad_interval=3(16dw), pad_amount=3(4dw)
    g1[1] = (int)((tensor_d0 & 0xFFFFu) << 16);            // tensor_dim0[15:0]
    g1[2] = (int)((tensor_d0 >> 16) | ((tensor_d1 & 0xFFFFu) << 16));
    g1[3] = (int)((tensor_d1 >> 16) | (tile_d0 << 16));    // tile_dim0
    g1[4] = (int)tile_d1;                                  // tile_dim1 (tile_dim2=0)
    g1[5] = (int)stride0;                                  // tensor_dim0_stride[31:0]
    g1[6] = 0;
    g1[7] = 0;
    i32x4 z4; z4[0] = 0; z4[1] = 0; z4[2] = 0; z4[3] = 0;
    i32x8 z8;
#pragma unroll
    for (int q = 0; q < 8; ++q) z8[q] = 0;
    __builtin_amdgcn_tensor_load_to_lds(g0, g1, z4, z4, z8, 0);
}

// ---------------------------------------------------------------------------
// Output projection: C[m=(t*B+b)][n] = A[m][:] @ WoutT[n][:] + bout[n]
// 128x128 block tile, 8 waves of 32x64, v_wmma_f32_16x16x32_bf16,
// double-buffered LDS fed by tensor_load_to_lds, TENSORcnt-pipelined.
// ---------------------------------------------------------------------------
__global__ __launch_bounds__(256)
void out_gemm(const __bf16* __restrict__ A, const __bf16* __restrict__ Bt,
              const float* __restrict__ bout, float* __restrict__ out) {
    constexpr int LDT = 40;                  // padded row stride (80B)
    __shared__ __bf16 sA[2][128 * LDT];
    __shared__ __bf16 sB[2][128 * LDT];

    const int tid   = threadIdx.x;
    const int lane  = tid & 31;
    const int wave  = tid >> 5;              // 0..7
    const int waveM = wave >> 1;             // 0..3 -> m offset waveM*32
    const int waveN = wave & 1;              // 0..1 -> n offset waveN*64
    const int n0 = blockIdx.x * 128;
    const int m0 = blockIdx.y * 128;
    const int l15 = lane & 15;

    v8f acc[2][4];
#pragma unroll
    for (int ms = 0; ms < 2; ++ms)
#pragma unroll
        for (int ns = 0; ns < 4; ++ns)
#pragma unroll
            for (int e = 0; e < 8; ++e) acc[ms][ns][e] = 0.f;

    // stage 0
    if (tid < 32) {
        tdm_load_2d((unsigned)(uintptr_t)&sA[0][0], A  + (size_t)m0 * H_,
                    H_, (unsigned)(B_ * T_), 32, 128, H_);
        tdm_load_2d((unsigned)(uintptr_t)&sB[0][0], Bt + (size_t)n0 * H_,
                    H_, (unsigned)V_, 32, 128, H_);
    }

    for (int ki = 0; ki < 16; ++ki) {
        const int buf = ki & 1;
        if (ki + 1 < 16) {
            if (tid < 32) {
                tdm_load_2d((unsigned)(uintptr_t)&sA[buf ^ 1][0],
                            A  + (size_t)m0 * H_ + (ki + 1) * 32,
                            H_, (unsigned)(B_ * T_), 32, 128, H_);
                tdm_load_2d((unsigned)(uintptr_t)&sB[buf ^ 1][0],
                            Bt + (size_t)n0 * H_ + (ki + 1) * 32,
                            H_, (unsigned)V_, 32, 128, H_);
                __builtin_amdgcn_s_wait_tensorcnt((short)2);  // stage ki done (in-order)
            }
        } else if (tid < 32) {
            __builtin_amdgcn_s_wait_tensorcnt((short)0);
        }
        __syncthreads();

        // A fragments: lanes 0-15 -> M=lane, K{0..7,16..23}; lanes 16-31 -> K{8..15,24..31}
        const int ksel = (lane < 16) ? 0 : 8;
        v16bf af[2];
#pragma unroll
        for (int ms = 0; ms < 2; ++ms) {
            const __bf16* p = &sA[buf][(waveM * 32 + ms * 16 + l15) * LDT];
            v8bf lo = *(const v8bf*)(p + ksel);
            v8bf hi = *(const v8bf*)(p + 16 + ksel);
#pragma unroll
            for (int e = 0; e < 8; ++e) { af[ms][e] = lo[e]; af[ms][8 + e] = hi[e]; }
        }
        // B fragments: lanes 0-15 -> N=lane, K{0..7,16..23}; lanes 16-31 -> K{8..15,24..31}
        const int koff = (lane < 16) ? 0 : 8;
        v16bf bfr[4];
#pragma unroll
        for (int ns = 0; ns < 4; ++ns) {
            const __bf16* p = &sB[buf][(waveN * 64 + ns * 16 + l15) * LDT + koff];
            v8bf lo = *(const v8bf*)(p);
            v8bf hi = *(const v8bf*)(p + 16);
#pragma unroll
            for (int e = 0; e < 8; ++e) { bfr[ns][e] = lo[e]; bfr[ns][8 + e] = hi[e]; }
        }

#pragma unroll
        for (int ms = 0; ms < 2; ++ms)
#pragma unroll
            for (int ns = 0; ns < 4; ++ns)
                acc[ms][ns] = __builtin_amdgcn_wmma_f32_16x16x32_bf16(
                    false, af[ms], false, bfr[ns], (short)0, acc[ms][ns],
                    false, false);

        __syncthreads();   // all reads of buf done before TDM overwrites it
    }

    // Epilogue: C 16x16 f32 layout (lanes 0-15: N=lane, M=v; lanes 16-31: M=8+v)
    const int mrow0 = (lane < 16) ? 0 : 8;
#pragma unroll
    for (int ns = 0; ns < 4; ++ns) {
        const int gn = n0 + waveN * 64 + ns * 16 + l15;
        const float bb = bout[gn];
#pragma unroll
        for (int ms = 0; ms < 2; ++ms) {
            const int mbase = m0 + waveM * 32 + ms * 16 + mrow0;
#pragma unroll
            for (int v = 0; v < 8; ++v) {
                int m = mbase + v;              // m = t*B + b
                int b = m & (B_ - 1);
                int t = m >> 5;
                out[((size_t)b * T_ + t) * V_ + gn] = acc[ms][ns][v] + bb;
            }
        }
    }
}

// ---------------------------------------------------------------------------
extern "C" void kernel_launch(void* const* d_in, const int* in_sizes, int n_in,
                              void* d_out, int out_size, void* d_ws, size_t ws_size,
                              hipStream_t stream) {
    const int*   wi   = (const int*)d_in[0];
    const int*   nw   = (const int*)d_in[1];
    const float* enc  = (const float*)d_in[2];
    const float* embW = (const float*)d_in[3];
    const float* Wg   = (const float*)d_in[4];
    const float* bgv  = (const float*)d_in[5];
    const float* Wc   = (const float*)d_in[6];
    const float* bcv  = (const float*)d_in[7];
    const float* Wout = (const float*)d_in[8];
    const float* bout = (const float*)d_in[9];
    float* out = (float*)d_out;

    char* ws = (char*)d_ws;
    size_t off = 0;
    auto carve = [&](size_t bytes) -> void* {
        void* p = ws + off;
        off += (bytes + 255) & ~(size_t)255;
        return p;
    };
    __bf16* WoutT = (__bf16*)carve((size_t)V_ * H_ * sizeof(__bf16));      // 32.8 MB [V][H]
    __bf16* Abf   = (__bf16*)carve((size_t)T_ * B_ * H_ * sizeof(__bf16)); // 2 MB [T*B][H]
    float*  gates = (float*)carve((size_t)B_ * G2H * sizeof(float));
    float*  h0    = (float*)carve((size_t)B_ * H_ * sizeof(float));
    float*  h1    = (float*)carve((size_t)B_ * H_ * sizeof(float));

    // Wout -> bf16 transposed [V][H]
    wout_transpose_bf16<<<dim3(V_ / 32, H_ / 32), 256, 0, stream>>>(Wout, WoutT);

    // Sequential GRU over T with h ping-pong
    float* hb[2] = { h0, h1 };
    for (int t = 0; t < T_; ++t) {
        const float* hin = (t == 0) ? enc : hb[(t - 1) & 1];
        float* hout = hb[t & 1];
        gru_gates<<<dim3(4, 4), 256, 0, stream>>>(wi, hin, embW, Wg, bgv, gates, t);
        gru_cand <<<dim3(2, 4), 256, 0, stream>>>(wi, nw, hin, embW, gates, Wc, bcv,
                                                  hout, Abf, t);
    }

    // Output projection with WMMA bf16, TDM-fed double-buffered LDS
    out_gemm<<<dim3(V_ / 128, (B_ * T_) / 128), 256, 0, stream>>>(Abf, WoutT, bout, out);
}
